// LinkPredictor_63599875719681
// MI455X (gfx1250) — matrix operand order repaired and checked by reading, not compiled, not run on hardware
//
#include <hip/hip_runtime.h>
#include <hip/hip_bf16.h>
#include <stdint.h>

#define HID     256
#define MBLK    128      // edges per tile
#define KCH     32       // K per chunk
#define NCHUNKS 16       // 512 / 32
#define NT      16       // N tiles (256/16)
#define ROWU    5        // lp_direct W row stride in 16B units
#define CHB     16384    // bytes per W chunk image (256 cols x 64B, operand-native)
#define CHU     1024     // 16B units per W chunk image
#define WBYTES  (NCHUNKS * CHB)          // 262144: bf16 W image in ws / LDS
#define ABYTES  (MBLK * ROWU * 16)       // 10240 per A buffer
#define LDS_RES (WBYTES + 2 * ABYTES)    // 282624 <= 320KB WGP LDS

typedef __attribute__((ext_vector_type(16))) __bf16 v16bf;
typedef __attribute__((ext_vector_type(8)))  __bf16 v8bf;
typedef __attribute__((ext_vector_type(8)))  float  v8f;

__device__ __forceinline__ v16bf cat16(v8bf lo, v8bf hi) {
    return __builtin_shufflevector(lo, hi, 0,1,2,3,4,5,6,7,8,9,10,11,12,13,14,15);
}
__device__ __forceinline__ v8bf cvt8(float4 a, float4 b) {
    v8bf r;
    r[0]=(__bf16)a.x; r[1]=(__bf16)a.y; r[2]=(__bf16)a.z; r[3]=(__bf16)a.w;
    r[4]=(__bf16)b.x; r[5]=(__bf16)b.y; r[6]=(__bf16)b.z; r[7]=(__bf16)b.w;
    return r;
}
__device__ __forceinline__ uint32_t lds_off(const void* p) {
    return (uint32_t)(size_t)p;          // LDS aperture: addr[31:0] = LDS offset
}
// CDNA5 async copy global -> LDS (GVS mode), tracked by ASYNCcnt
__device__ __forceinline__ void async_b128(uint32_t ldsaddr, uint32_t voff, uint64_t sbase) {
    asm volatile("global_load_async_to_lds_b128 %0, %1, %2 offset:0"
                 :: "v"(ldsaddr), "v"(voff), "s"(sbase) : "memory");
}
__device__ __forceinline__ void wait_async0() {
    asm volatile("s_wait_asynccnt 0x0" ::: "memory");
}

// ---- one-time: W1 f32 [512][256] -> bf16 operand-native image in ws ----
// layout: byte off = c*16384 + t*1024 + g*512 + nl*32 + (k%16)*2
//   (c = k/32 chunk, t = n/16 tile, g = (k%32)/16 lane-half, nl = n%16)
__global__ __launch_bounds__(256)
void w_preconvert(const float* __restrict__ W1, uint32_t* __restrict__ ws) {
    const int id = blockIdx.x * 256 + threadIdx.x;   // 65536 threads
    const int kk = id >> 8;                          // k-pair 0..255 (k = 2*kk)
    const int n  = id & 255;
    const int c  = kk >> 4;
    const int kl = kk & 15;                          // pair index within chunk
    const int g  = kl >> 3;
    const int t  = n >> 4;
    const int nl = n & 15;
    const float a = W1[(2 * kk    ) * HID + n];
    const float b = W1[(2 * kk + 1) * HID + n];
    const uint16_t ua = __builtin_bit_cast(uint16_t, (__bf16)a);
    const uint16_t ub = __builtin_bit_cast(uint16_t, (__bf16)b);
    ws[c * 4096 + t * 256 + g * 128 + nl * 8 + (kl & 7)] =
        (uint32_t)ua | ((uint32_t)ub << 16);
}

// ---- shared epilogue: bias + ReLU + dot(W2) + reduce + store ----
__device__ __forceinline__ void epilogue(const v8f* acc, const float* __restrict__ b1,
                                         const float* __restrict__ W2,
                                         const float* __restrict__ b2,
                                         float* __restrict__ out, int E, int e0,
                                         int wave, int nrow, int hh) {
    float s[8];
#pragma unroll
    for (int r = 0; r < 8; ++r) s[r] = 0.f;
#pragma unroll
    for (int t = 0; t < NT; ++t) {
        const float bb = b1[t * 16 + nrow];
        const float w2 = W2[t * 16 + nrow];
#pragma unroll
        for (int r = 0; r < 8; ++r) {
            float h = acc[t][r] + bb;
            h = h > 0.f ? h : 0.f;
            s[r] += h * w2;
        }
    }
#pragma unroll
    for (int r = 0; r < 8; ++r) {
        float v = s[r];
        v += __shfl_xor(v, 1, 32);
        v += __shfl_xor(v, 2, 32);
        v += __shfl_xor(v, 4, 32);
        v += __shfl_xor(v, 8, 32);       // stays within each 16-lane half
        s[r] = v;
    }
    if (nrow == 0) {
        const float bias2 = b2[0];
        const int ebase = e0 + wave * 16 + hh * 8;
#pragma unroll
        for (int r = 0; r < 8; ++r) {
            const int e = ebase + r;
            if (e < E) out[e] = s[r] + bias2;
        }
    }
}

// ---- primary: resident bf16 W1 (256KB) in LDS, persistent tiles ----
__global__ __launch_bounds__(256)
void lp_resident(const float* __restrict__ emb_src, const float* __restrict__ emb_dst,
                 const int* __restrict__ eidx, const uint32_t* __restrict__ wsW,
                 const float* __restrict__ b1, const float* __restrict__ W2,
                 const float* __restrict__ b2, float* __restrict__ out,
                 int E, int ntiles) {
    extern __shared__ char smem[];
    const v8bf* Wv = (const v8bf*)smem;                    // operand-native image
    v8bf* const A0 = (v8bf*)(smem + WBYTES);
    v8bf* const A1 = (v8bf*)(smem + WBYTES + ABYTES);

    const int tid  = threadIdx.x;
    const int lane = tid & 31, wave = tid >> 5;
    const int nrow = lane & 15, hh = lane >> 4;
    const int le   = tid >> 1,  lp = tid & 1;
    const int ub   = hh * 32 + nrow * 2;                   // lane's B unit offset

    // ---- load full bf16 W1 image into LDS once via async DMA ----
    {
        const uint64_t g  = (uint64_t)wsW;
        const uint32_t lb = lds_off(smem);
#pragma unroll 1
        for (int j = 0; j < 64; ++j) {                     // 64 x 256 x 16B = 256KB
            const uint32_t off = (uint32_t)(tid + 256 * j) * 16;
            async_b128(lb + off, off, g);
        }
        wait_async0();
        __syncthreads();
    }

    for (int tile = blockIdx.x; tile < ntiles; tile += gridDim.x) {
        const int e0 = tile * MBLK;
        int ge = e0 + le; if (ge >= E) ge = E - 1;         // clamp tail, stores masked
        const int si = eidx[ge];
        const int di = eidx[E + ge];

        // stage chunk 0 (src table) into buffer 0
        {
            const float4* g = (const float4*)(emb_src + (size_t)si * HID + lp * 16);
            float4 f0 = g[0], f1 = g[1], f2 = g[2], f3 = g[3];
            A0[le * ROWU + 2 * lp + 0] = cvt8(f0, f1);
            A0[le * ROWU + 2 * lp + 1] = cvt8(f2, f3);
        }
        __syncthreads();

        v8f zf = {};
        v8f acc[NT];
#pragma unroll
        for (int t = 0; t < NT; ++t) acc[t] = zf;

        for (int c = 0; c < NCHUNKS; ++c) {
            // issue gather for chunk c+1 early (latency hidden behind WMMAs)
            float4 f0, f1, f2, f3;
            const bool have = (c + 1 < NCHUNKS);
            if (have) {
                const float* sp = (c + 1 < 8)
                    ? (emb_src + (size_t)si * HID + (c + 1) * KCH)
                    : (emb_dst + (size_t)di * HID + (c + 1 - 8) * KCH);
                const float4* g = (const float4*)(sp + lp * 16);
                f0 = g[0]; f1 = g[1]; f2 = g[2]; f3 = g[3];
            }

            // compute chunk c: 16 WMMAs, B tiles one ahead, all DS immediates
            const v8bf* Ab = (c & 1) ? A1 : A0;
            const int arow = wave * 16 + nrow;
            v16bf a = cat16(Ab[arow * ROWU + 2 * hh], Ab[arow * ROWU + 2 * hh + 1]);

            const int u = c * CHU + ub;                    // per-chunk base (uniform add)
            v8bf blo = Wv[u], bhi = Wv[u + 1];
#pragma unroll
            for (int t = 0; t < NT; ++t) {
                v8bf nlo, nhi;
                if (t < NT - 1) { nlo = Wv[u + (t + 1) * 64]; nhi = Wv[u + (t + 1) * 64 + 1]; }
                v16bf b = cat16(blo, bhi);
                acc[t] = __builtin_amdgcn_wmma_f32_16x16x32_bf16(
                             false, a, false, b, (short)0, acc[t], false, false);
                blo = nlo; bhi = nhi;
            }

            // convert + store staged chunk c+1
            if (have) {
                v8bf* dst = ((c + 1) & 1) ? A1 : A0;
                dst[le * ROWU + 2 * lp + 0] = cvt8(f0, f1);
                dst[le * ROWU + 2 * lp + 1] = cvt8(f2, f3);
            }
            __syncthreads();
        }
        epilogue(acc, b1, W2, b2, out, E, e0, wave, nrow, hh);
    }
}

// ---- fallback: per-chunk double-buffered staging, 53KB static LDS ----
__global__ __launch_bounds__(256)
void lp_chunked(const float* __restrict__ emb_src, const float* __restrict__ emb_dst,
                const int* __restrict__ eidx, const uint32_t* __restrict__ wsW,
                const float* __restrict__ b1, const float* __restrict__ W2,
                const float* __restrict__ b2, float* __restrict__ out, int E) {
    __shared__ v8bf Wb[2][CHU];           // 2 x 16384B
    __shared__ v8bf Ab2[2][MBLK * ROWU];  // 2 x 10240B

    const int tid  = threadIdx.x;
    const int lane = tid & 31, wave = tid >> 5;
    const int nrow = lane & 15, hh = lane >> 4;
    const int le   = tid >> 1,  lp = tid & 1;
    const int e0   = blockIdx.x * MBLK;
    const int ub   = hh * 32 + nrow * 2;

    int ge = e0 + le; if (ge >= E) ge = E - 1;
    const int si = eidx[ge];
    const int di = eidx[E + ge];
    const uint64_t wg = (uint64_t)wsW;

    // stage chunk 0: A gather + async W copy (contiguous 16KB slab)
    {
        const float4* g = (const float4*)(emb_src + (size_t)si * HID + lp * 16);
        float4 f0 = g[0], f1 = g[1], f2 = g[2], f3 = g[3];
        Ab2[0][le * ROWU + 2 * lp + 0] = cvt8(f0, f1);
        Ab2[0][le * ROWU + 2 * lp + 1] = cvt8(f2, f3);
        const uint32_t lb = lds_off(&Wb[0][tid * 4]);
#pragma unroll
        for (int sub = 0; sub < 4; ++sub)
            async_b128(lb + sub * 16, (uint32_t)(tid * 64 + sub * 16), wg);
    }
    wait_async0();
    __syncthreads();

    v8f zf = {};
    v8f acc[NT];
#pragma unroll
    for (int t = 0; t < NT; ++t) acc[t] = zf;

    for (int c = 0; c < NCHUNKS; ++c) {
        float4 f0, f1, f2, f3;
        const bool have = (c + 1 < NCHUNKS);
        if (have) {
            const float* sp = (c + 1 < 8)
                ? (emb_src + (size_t)si * HID + (c + 1) * KCH)
                : (emb_dst + (size_t)di * HID + (c + 1 - 8) * KCH);
            const float4* g = (const float4*)(sp + lp * 16);
            f0 = g[0]; f1 = g[1]; f2 = g[2]; f3 = g[3];
            const uint32_t lb = lds_off(&Wb[(c + 1) & 1][tid * 4]);
#pragma unroll
            for (int sub = 0; sub < 4; ++sub)
                async_b128(lb + sub * 16,
                           (uint32_t)((c + 1) * CHB + tid * 64 + sub * 16), wg);
        }

        const v8bf* Ab = Ab2[c & 1];
        const v8bf* Wc = Wb[c & 1];
        const int arow = wave * 16 + nrow;
        v16bf a = cat16(Ab[arow * ROWU + 2 * hh], Ab[arow * ROWU + 2 * hh + 1]);

        v8bf blo = Wc[ub], bhi = Wc[ub + 1];
#pragma unroll
        for (int t = 0; t < NT; ++t) {
            v8bf nlo, nhi;
            if (t < NT - 1) { nlo = Wc[ub + (t + 1) * 64]; nhi = Wc[ub + (t + 1) * 64 + 1]; }
            v16bf b = cat16(blo, bhi);
            acc[t] = __builtin_amdgcn_wmma_f32_16x16x32_bf16(
                         false, a, false, b, (short)0, acc[t], false, false);
            blo = nlo; bhi = nhi;
        }

        if (have) {
            v8bf* dst = Ab2[(c + 1) & 1];
            dst[le * ROWU + 2 * lp + 0] = cvt8(f0, f1);
            dst[le * ROWU + 2 * lp + 1] = cvt8(f2, f3);
        }
        wait_async0();
        __syncthreads();
    }
    epilogue(acc, b1, W2, b2, out, E, e0, wave, nrow, hh);
}

// ---- ultra-fallback: no ws needed ----
__global__ __launch_bounds__(256)
void lp_direct(const float* __restrict__ emb_src, const float* __restrict__ emb_dst,
               const int* __restrict__ eidx, const float* __restrict__ W1,
               const float* __restrict__ b1, const float* __restrict__ W2,
               const float* __restrict__ b2, float* __restrict__ out, int E) {
    __shared__ v8bf ldsA[MBLK * ROWU];
    __shared__ v8bf ldsW[HID * ROWU];

    const int tid  = threadIdx.x;
    const int lane = tid & 31, wave = tid >> 5;
    const int nrow = lane & 15, hh = lane >> 4;
    const int le   = tid >> 1,  lp = tid & 1;
    const int e0   = blockIdx.x * MBLK;

    int ge = e0 + le; if (ge >= E) ge = E - 1;
    const int si = eidx[ge];
    const int di = eidx[E + ge];

    v8f zf = {};
    v8f acc[NT];
#pragma unroll
    for (int t = 0; t < NT; ++t) acc[t] = zf;

    for (int c = 0; c < NCHUNKS; ++c) {
        const float* srcp = (c < 8)
            ? (emb_src + (size_t)si * HID + c * KCH)
            : (emb_dst + (size_t)di * HID + (c - 8) * KCH);
        const float4* g = (const float4*)(srcp + lp * 16);
        float4 f0 = g[0], f1 = g[1], f2 = g[2], f3 = g[3];

        __syncthreads();
        ldsA[le * ROWU + 2 * lp + 0] = cvt8(f0, f1);
        ldsA[le * ROWU + 2 * lp + 1] = cvt8(f2, f3);

        const float* wp = W1 + (size_t)c * KCH * HID + tid;
#pragma unroll
        for (int grp = 0; grp < 4; ++grp) {
            v8bf wv;
#pragma unroll
            for (int j = 0; j < 8; ++j) wv[j] = (__bf16)wp[(grp * 8 + j) * HID];
            ldsW[tid * ROWU + grp] = wv;
        }
        __syncthreads();

        const int arow = wave * 16 + nrow;
        v16bf a = cat16(ldsA[arow * ROWU + 2 * hh], ldsA[arow * ROWU + 2 * hh + 1]);
        const int base = nrow * ROWU + 2 * hh;
        v8bf blo = ldsW[base], bhi = ldsW[base + 1];
#pragma unroll
        for (int t = 0; t < NT; ++t) {
            v16bf b = cat16(blo, bhi);
            if (t < NT - 1) {
                const int nb = base + (t + 1) * 16 * ROWU;
                blo = ldsW[nb]; bhi = ldsW[nb + 1];
            }
            acc[t] = __builtin_amdgcn_wmma_f32_16x16x32_bf16(
                         false, a, false, b, (short)0, acc[t], false, false);
        }
    }
    epilogue(acc, b1, W2, b2, out, E, e0, wave, nrow, hh);
}

extern "C" void kernel_launch(void* const* d_in, const int* in_sizes, int n_in,
                              void* d_out, int out_size, void* d_ws, size_t ws_size,
                              hipStream_t stream) {
    const float* emb_src = (const float*)d_in[0];
    const float* emb_dst = (const float*)d_in[1];
    const int*   eidx    = (const int*)  d_in[2];
    const float* W1      = (const float*)d_in[3];
    const float* b1      = (const float*)d_in[4];
    const float* W2      = (const float*)d_in[5];
    const float* b2      = (const float*)d_in[6];
    float* out = (float*)d_out;

    const int E = in_sizes[2] / 2;
    const int ntiles = (E + MBLK - 1) / MBLK;

    if (ws_size < (size_t)WBYTES) {
        hipLaunchKernelGGL(lp_direct, dim3(ntiles), dim3(256), 0, stream,
                           emb_src, emb_dst, eidx, W1, b1, W2, b2, out, E);
        return;
    }

    uint32_t* ws32 = (uint32_t*)d_ws;
    hipLaunchKernelGGL(w_preconvert, dim3(256), dim3(256), 0, stream, W1, ws32);

    int dev = 0, maxsh = 0;
    (void)hipGetDevice(&dev);
    (void)hipDeviceGetAttribute(&maxsh, hipDeviceAttributeMaxSharedMemoryPerBlock, dev);

    if (maxsh >= LDS_RES) {
        const int grid = ntiles < 512 ? ntiles : 512;
        hipLaunchKernelGGL(lp_resident, dim3(grid), dim3(256), LDS_RES, stream,
                           emb_src, emb_dst, eidx, ws32, b1, W2, b2, out, E, ntiles);
    } else {
        hipLaunchKernelGGL(lp_chunked, dim3(ntiles), dim3(256), 0, stream,
                           emb_src, emb_dst, eidx, ws32, b1, W2, b2, out, E);
    }
}
